// FeatureCrossAttention_4088808866414
// MI455X (gfx1250) — compile-verified, matrix operand order checked by reading, not run on hardware
//
#include <hip/hip_runtime.h>

// ---------- CDNA5 wave32 WMMA types ----------
typedef __attribute__((ext_vector_type(16))) __bf16 v16bf;
typedef __attribute__((ext_vector_type(8)))  float  v8f;

union BF16x16 {
  v16bf v;
  unsigned short s[16];
  uint4 q[2];
};

// native f32 -> bf16 (RNE) via hardware cvt
__device__ __forceinline__ unsigned short f2bf(float f) {
  __bf16 h = (__bf16)f;
  return __builtin_bit_cast(unsigned short, h);
}

// 16 bf16 (32B) contiguous from global -> A/B fragment
__device__ __forceinline__ v16bf load_bf16x16(const unsigned short* p) {
  BF16x16 r;
  r.q[0] = *(const uint4*)(p);
  r.q[1] = *(const uint4*)(p + 8);
  return r.v;
}

// 16 f32 (64B) contiguous from global, convert -> fragment (native cvt)
__device__ __forceinline__ v16bf load_f32_as_bf16x16(const float* p) {
  const float4* p4 = (const float4*)p;
  float4 f0 = p4[0], f1 = p4[1], f2 = p4[2], f3 = p4[3];
  v16bf r;
  r[0]  = (__bf16)f0.x; r[1]  = (__bf16)f0.y; r[2]  = (__bf16)f0.z; r[3]  = (__bf16)f0.w;
  r[4]  = (__bf16)f1.x; r[5]  = (__bf16)f1.y; r[6]  = (__bf16)f1.z; r[7]  = (__bf16)f1.w;
  r[8]  = (__bf16)f2.x; r[9]  = (__bf16)f2.y; r[10] = (__bf16)f2.z; r[11] = (__bf16)f2.w;
  r[12] = (__bf16)f3.x; r[13] = (__bf16)f3.y; r[14] = (__bf16)f3.z; r[15] = (__bf16)f3.w;
  return r;
}

#define WMMA_BF16(A, B, C) \
  __builtin_amdgcn_wmma_f32_16x16x32_bf16(false, (A), false, (B), (short)0, (C), false, false)

// Problem constants
#define BATCH 16
#define SEQ   4096
#define DIN   512
#define CPROJ 128
#define MOUT  512
#define ROWS  (BATCH * SEQ)        // 65536
#define YSIZE (ROWS * MOUT)        // 33554432

// ---------------------------------------------------------------------------
// Kernel 0: weight transpose + f32->bf16.   WT[n][k] layouts (K contiguous).
// blockIdx.y: 0..2 -> WqT/WkT/WvT (128x512), 3 -> WqoT (512x128)
// ---------------------------------------------------------------------------
__global__ __launch_bounds__(256) void prep_kernel(
    const float* __restrict__ Wq, const float* __restrict__ Wk,
    const float* __restrict__ Wv, const float* __restrict__ Wqo,
    unsigned short* __restrict__ WqT, unsigned short* __restrict__ WkT,
    unsigned short* __restrict__ WvT, unsigned short* __restrict__ WqoT)
{
  const int which = blockIdx.y;
  const int g = blockIdx.x * 256 + threadIdx.x;     // < 65536
  if (which < 3) {
    const float* W = (which == 0) ? Wq : (which == 1) ? Wk : Wv;
    unsigned short* O = (which == 0) ? WqT : (which == 1) ? WkT : WvT;
    const int n = g >> 9, k = g & 511;              // O[n][k] = W[k][n]
    O[g] = f2bf(W[k * CPROJ + n]);
  } else {
    const int m = g >> 7, k = g & 127;              // WqoT[m][k] = Wqo[k][m]
    WqoT[g] = f2bf(Wqo[k * MOUT + m]);
  }
}

// ---------------------------------------------------------------------------
// Kernel 1: fused QKV projection.  128 rows per block; each of the 8 waves
// owns a 16-row M-tile and computes all 8 N-tiles (full 128 output cols).
// fp32 A fragment loaded+converted ONCE per wave per K-step; all 8 bf16 B
// fragments are materialized BEFORE the WMMA burst so loads issue as one
// clause and overlap WMMA execution (no per-WMMA s_wait_loadcnt 0).
// blockIdx.y: 0 -> Qm (normal + transposed), 1 -> Km (transposed), 2 -> Vm (normal)
// Normal:  bf16 (B*L, 128).  Transposed: bf16 (B, 128, L).
// ---------------------------------------------------------------------------
__global__ __launch_bounds__(256) void proj_kernel(
    const float* __restrict__ Qin, const float* __restrict__ KVin,
    const unsigned short* __restrict__ WqT, const unsigned short* __restrict__ WkT,
    const unsigned short* __restrict__ WvT,
    const float* __restrict__ bq, const float* __restrict__ bk,
    const float* __restrict__ bv,
    unsigned short* __restrict__ QmN, unsigned short* __restrict__ VmN,
    unsigned short* __restrict__ QmT, unsigned short* __restrict__ KmT)
{
  __shared__ unsigned short lds[8 * 16 * 136];      // per-wave 16x128 tile, padded

  const int which = blockIdx.y;
  const float* X = (which == 0) ? Qin : KVin;
  const unsigned short* WT = (which == 0) ? WqT : (which == 1) ? WkT : WvT;
  const float* bias = (which == 0) ? bq : (which == 1) ? bk : bv;
  unsigned short* outN = (which == 0) ? QmN : (which == 2) ? VmN : nullptr;
  unsigned short* outT = (which == 0) ? QmT : (which == 1) ? KmT : nullptr;

  const int tid = threadIdx.x;
  const int lane = tid & 31;
  const int w = tid >> 5;              // wave id = M tile (0..7)
  const int l16 = lane & 15;
  const int lhalf = lane >> 4;

  const int wrow = blockIdx.x * 128 + w * 16;   // wave's row base (B*L flattened)
  const int arow = wrow + l16;                  // A row for this lane
  const unsigned short* Wlane = WT + (size_t)l16 * DIN;  // n-tile stride 16*512

  v8f acc[8] = {{}, {}, {}, {}, {}, {}, {}, {}};
#pragma unroll 2
  for (int kk = 0; kk < 16; ++kk) {
    const int k = kk * 32 + lhalf * 16;
    // issue all B loads + the A load up front, then burst 8 WMMAs
    v16bf bfr[8];
#pragma unroll
    for (int n = 0; n < 8; ++n)
      bfr[n] = load_bf16x16(Wlane + (size_t)n * (16 * DIN) + k);
    v16bf a = load_f32_as_bf16x16(X + (size_t)arow * DIN + k);
#pragma unroll
    for (int n = 0; n < 8; ++n)
      acc[n] = WMMA_BF16(a, bfr[n], acc[n]);
  }

  const int bidx = wrow >> 12;                  // batch (4096 % 128 == 0)
  const int lloc = (wrow & 4095) + lhalf * 8;   // seq offset for T-store
  unsigned short* ldsw = lds + w * (16 * 136);  // this wave's staging tile

#pragma unroll
  for (int n = 0; n < 8; ++n) {
    const int ncol = n * 16 + l16;              // output column for this lane
    const float bval = bias[ncol];
    unsigned short obf[8];
#pragma unroll
    for (int r = 0; r < 8; ++r) obf[r] = f2bf(acc[n][r] + bval);

    if (outT) {
      // lane holds col c=ncol, 8 consecutive sequence rows -> one 16B store
      union { uint4 q; unsigned short s[8]; } pk;
#pragma unroll
      for (int r = 0; r < 8; ++r) pk.s[r] = obf[r];
      *(uint4*)(outT + ((size_t)bidx * CPROJ + ncol) * SEQ + lloc) = pk.q;
    }
    if (outN) {
#pragma unroll
      for (int r = 0; r < 8; ++r)
        ldsw[(lhalf * 8 + r) * 136 + ncol] = obf[r];
    }
  }

  if (outN) {
    // wave-private staging: only in-wave LDS ordering required (no barrier)
    const int row = lane >> 1;                  // 0..15
    const int c8 = (lane & 1) * 64;             // two 64-col halves
#pragma unroll
    for (int j = 0; j < 8; ++j) {
      uint4 pk = *(const uint4*)(&ldsw[row * 136 + c8 + j * 8]);
      *(uint4*)(outN + (size_t)(wrow + row) * CPROJ + c8 + j * 8) = pk;
    }
  }
}

// ---------------------------------------------------------------------------
// Kernel 2: scores[b,h,d,e] = SCALE * sum_l Qh[d,l]*Kh[e,l]
// One block per (b,h): 4 waves = 2x2 16x16 tiles, K loop over L=4096.
// 4 K-steps of operands are materialized before each 4-WMMA burst.
// ---------------------------------------------------------------------------
__global__ __launch_bounds__(128) void scores_kernel(
    const unsigned short* __restrict__ QmT, const unsigned short* __restrict__ KmT,
    float* __restrict__ scores)
{
  const int bh = blockIdx.x;                 // 0..63
  const int b = bh >> 2, h = bh & 3;
  const int tid = threadIdx.x, lane = tid & 31, w = tid >> 5;
  const int mt = w >> 1, nt = w & 1;
  const int l16 = lane & 15, lhalf = lane >> 4;

  const unsigned short* Ab = QmT + ((size_t)b * CPROJ + h * 32 + mt * 16 + l16) * SEQ;
  const unsigned short* Bb = KmT + ((size_t)b * CPROJ + h * 32 + nt * 16 + l16) * SEQ;

  v8f acc = {};
  for (int kk = 0; kk < SEQ / 32; kk += 4) {
    v16bf a[4], bfr[4];
#pragma unroll
    for (int j = 0; j < 4; ++j) {
      const int k = (kk + j) * 32 + lhalf * 16;
      a[j]   = load_bf16x16(Ab + k);
      bfr[j] = load_bf16x16(Bb + k);
    }
#pragma unroll
    for (int j = 0; j < 4; ++j)
      acc = WMMA_BF16(a[j], bfr[j], acc);
  }

  const float SCALE = 0.1767766952966369f;   // 1/sqrt(32)
  float* S = scores + (size_t)bh * 1024;
  const int e = nt * 16 + l16;
#pragma unroll
  for (int r = 0; r < 8; ++r) {
    const int d = mt * 16 + lhalf * 8 + r;
    S[d * 32 + e] = acc[r] * SCALE;
  }
}

// ---------------------------------------------------------------------------
// Kernel 3: row softmax over 32 elements; one wave per row (wave32).
// Writes attn directly into d_out's second segment.
// ---------------------------------------------------------------------------
__global__ __launch_bounds__(256) void softmax_kernel(
    const float* __restrict__ scores, float* __restrict__ attn)
{
  const int tid = threadIdx.x, lane = tid & 31, w = tid >> 5;
  const int row = blockIdx.x * 8 + w;        // 0..2047
  float v = scores[row * 32 + lane];
  float m = v;
#pragma unroll
  for (int off = 16; off; off >>= 1) m = fmaxf(m, __shfl_xor(m, off, 32));
  float e = __expf(v - m);
  float s = e;
#pragma unroll
  for (int off = 16; off; off >>= 1) s += __shfl_xor(s, off, 32);
  attn[row * 32 + lane] = e / s;
}

// ---------------------------------------------------------------------------
// Kernel 4: per-batch effective output weight (transposed, bf16):
//   W2T[b][m][h*32+e] = sum_d attn[b,h,d,e] * Wo[h*32+d][m]
// ---------------------------------------------------------------------------
__global__ __launch_bounds__(256) void woeff_kernel(
    const float* __restrict__ attn, const float* __restrict__ Wo,
    unsigned short* __restrict__ W2T)
{
  const int g = blockIdx.x * 256 + threadIdx.x;    // < 16*128*512
  const int b = g >> 16;
  const int rem = g & 65535;
  const int c = rem >> 9;        // 0..127  (h*32+e)
  const int m = rem & 511;
  const int h = c >> 5, e = c & 31;
  const float* A = attn + (size_t)((b * 4 + h) * 32) * 32 + e;
  const float* W = Wo + (size_t)(h * 32) * MOUT + m;
  float s = 0.f;
#pragma unroll 8
  for (int d = 0; d < 32; ++d) s += A[d * 32] * W[d * MOUT];
  W2T[((size_t)b * MOUT + m) * CPROJ + c] = f2bf(s);
}

// ---------------------------------------------------------------------------
// Kernel 5: fused output GEMM + residual bias + LayerNorm.
//   x[l][m] = sum_{c<128} Qm[l][c]*Wqo[c][m] + sum_{c<128} Vm[l][c]*Wo_eff[b][c][m]
//             + bo[m] + bqo[m];  y = LN(x)*gamma + beta
// Computed as x^T tiles: A = weight^T (K contig), B = Qm/Vm rows (K contig).
// 16 seq rows per block; 8 waves x 4 M-tiles = 512 features; K = 256 (8 steps).
// All 4 A fragments materialized before the 4-WMMA burst.
// ---------------------------------------------------------------------------
__global__ __launch_bounds__(256) void out_ln_kernel(
    const unsigned short* __restrict__ QmN, const unsigned short* __restrict__ VmN,
    const unsigned short* __restrict__ WqoT, const unsigned short* __restrict__ W2T,
    const float* __restrict__ bo, const float* __restrict__ bqo,
    const float* __restrict__ gamma, const float* __restrict__ beta,
    float* __restrict__ y)
{
  __shared__ float xs[16 * 516];             // padded stride vs LDS banks
  __shared__ float biass[512];
  __shared__ float mus[16], rss[16];

  const int tid = threadIdx.x, lane = tid & 31, w = tid >> 5;
  const int l16 = lane & 15, lhalf = lane >> 4;
  const int grow = blockIdx.x * 16;          // flattened (B*L) row base
  const int b = grow >> 12;

  biass[tid] = bo[tid] + bqo[tid];
  biass[tid + 256] = bo[tid + 256] + bqo[tid + 256];

  const unsigned short* Bq = QmN + (size_t)(grow + l16) * CPROJ;
  const unsigned short* Bv = VmN + (size_t)(grow + l16) * CPROJ;
  const unsigned short* W2b = W2T + (size_t)b * MOUT * CPROJ;

  v8f acc[4] = {{}, {}, {}, {}};
#pragma unroll
  for (int kk = 0; kk < 8; ++kk) {
    const int k = kk * 32 + lhalf * 16;      // 0..255
    const unsigned short* In;
    const unsigned short* Wt;
    int ko;
    if (k < CPROJ) { In = Bq; Wt = WqoT; ko = k; }
    else           { In = Bv; Wt = W2b;  ko = k - CPROJ; }
    v16bf bfrag = load_bf16x16(In + ko);
    v16bf afrag[4];
#pragma unroll
    for (int f = 0; f < 4; ++f) {
      const int m = (w * 4 + f) * 16 + l16;
      afrag[f] = load_bf16x16(Wt + (size_t)m * CPROJ + ko);
    }
#pragma unroll
    for (int f = 0; f < 4; ++f)
      acc[f] = WMMA_BF16(afrag[f], bfrag, acc[f]);
  }
  __syncthreads();                           // biass ready

  // D tile is x^T: lane = seq col l16, rows = 8 consecutive features
#pragma unroll
  for (int f = 0; f < 4; ++f) {
    const int mbase = (w * 4 + f) * 16 + lhalf * 8;
#pragma unroll
    for (int r = 0; r < 8; ++r)
      xs[l16 * 516 + mbase + r] = acc[f][r] + biass[mbase + r];
  }
  __syncthreads();

  // LayerNorm: 8 waves x (2 half-waves) cover the 16 rows; width-16 shuffles
  const int row = w * 2 + lhalf;
  float s1 = 0.f, s2 = 0.f;
#pragma unroll 8
  for (int j = 0; j < 32; ++j) {
    float x = xs[row * 516 + l16 + 16 * j];
    s1 += x; s2 += x * x;
  }
#pragma unroll
  for (int off = 8; off; off >>= 1) {
    s1 += __shfl_xor(s1, off, 16);
    s2 += __shfl_xor(s2, off, 16);
  }
  if (l16 == 0) {
    const float mu = s1 * (1.f / 512.f);
    const float var = s2 * (1.f / 512.f) - mu * mu;
    mus[row] = mu;
    rss[row] = rsqrtf(var + 1e-5f);
  }
  __syncthreads();

#pragma unroll
  for (int j = 0; j < 32; ++j) {
    const int idx = tid + 256 * j;           // 0..8191
    const int r2 = idx >> 9, cc = idx & 511;
    const float xv = xs[r2 * 516 + cc];
    y[(size_t)(grow + r2) * MOUT + cc] =
        (xv - mus[r2]) * rss[r2] * gamma[cc] + beta[cc];
  }
}

// ---------------------------------------------------------------------------
extern "C" void kernel_launch(void* const* d_in, const int* in_sizes, int n_in,
                              void* d_out, int out_size, void* d_ws, size_t ws_size,
                              hipStream_t stream) {
  (void)in_sizes; (void)n_in; (void)out_size; (void)ws_size;

  const float* Q_in  = (const float*)d_in[0];
  const float* KV_in = (const float*)d_in[1];
  const float* Wq    = (const float*)d_in[2];
  const float* bq    = (const float*)d_in[3];
  const float* Wk    = (const float*)d_in[4];
  const float* bk    = (const float*)d_in[5];
  const float* Wv    = (const float*)d_in[6];
  const float* bv    = (const float*)d_in[7];
  const float* Wo    = (const float*)d_in[8];
  const float* bo    = (const float*)d_in[9];
  const float* Wqo   = (const float*)d_in[10];
  const float* bqo   = (const float*)d_in[11];
  const float* gamma = (const float*)d_in[12];
  const float* beta  = (const float*)d_in[13];

  float* y    = (float*)d_out;
  float* attn = y + (size_t)YSIZE;

  // workspace carve-up (256B aligned)
  char* ws = (char*)d_ws;
  size_t off = 0;
  auto carve = [&](size_t bytes) {
    void* p = ws + off;
    off += (bytes + 255) & ~(size_t)255;
    return p;
  };
  unsigned short* WqT  = (unsigned short*)carve(CPROJ * DIN * 2);
  unsigned short* WkT  = (unsigned short*)carve(CPROJ * DIN * 2);
  unsigned short* WvT  = (unsigned short*)carve(CPROJ * DIN * 2);
  unsigned short* WqoT = (unsigned short*)carve(MOUT * CPROJ * 2);
  unsigned short* W2T  = (unsigned short*)carve((size_t)BATCH * MOUT * CPROJ * 2);
  float*          scoresW = (float*)carve((size_t)BATCH * 4 * 32 * 32 * 4);
  unsigned short* QmN = (unsigned short*)carve((size_t)ROWS * CPROJ * 2);
  unsigned short* VmN = (unsigned short*)carve((size_t)ROWS * CPROJ * 2);
  unsigned short* QmT = (unsigned short*)carve((size_t)ROWS * CPROJ * 2);
  unsigned short* KmT = (unsigned short*)carve((size_t)ROWS * CPROJ * 2);

  prep_kernel<<<dim3(256, 4), 256, 0, stream>>>(Wq, Wk, Wv, Wqo,
                                                WqT, WkT, WvT, WqoT);
  proj_kernel<<<dim3(ROWS / 128, 3), 256, 0, stream>>>(
      Q_in, KV_in, WqT, WkT, WvT, bq, bk, bv, QmN, VmN, QmT, KmT);
  scores_kernel<<<BATCH * 4, 128, 0, stream>>>(QmT, KmT, scoresW);
  softmax_kernel<<<256, 256, 0, stream>>>(scoresW, attn);
  woeff_kernel<<<4096, 256, 0, stream>>>(attn, Wo, W2T);
  out_ln_kernel<<<ROWS / 16, 256, 0, stream>>>(QmN, VmN, WqoT, W2T,
                                               bo, bqo, gamma, beta, y);
}
